// WanModel_15676630630460
// MI455X (gfx1250) — compile-verified, matrix operand order checked by reading, not run on hardware
//
#include <hip/hip_runtime.h>
#include <hip/hip_bf16.h>
#include <math.h>

// ---------------------------------------------------------------- types
typedef __attribute__((ext_vector_type(16))) __bf16 v16bf;
typedef __attribute__((ext_vector_type(8)))  float  v8f;
typedef __attribute__((ext_vector_type(8)))  unsigned int v8u;
typedef __attribute__((ext_vector_type(4)))  unsigned int v4u;
typedef __attribute__((ext_vector_type(4)))  int    v4i;
typedef __attribute__((ext_vector_type(8)))  int    v8i;
typedef unsigned short ushort_t;

#define TM      128   // block tile M
#define TN      128   // block tile N
#define KSTEP   32
#define ALDS_E  34    // padded bf16 LDS row stride (32 + 1 dword pad)

__device__ __forceinline__ ushort_t f2bf_us(float f) {
  unsigned int u = __builtin_bit_cast(unsigned int, f);
  u += 0x7FFFu + ((u >> 16) & 1u);          // round-to-nearest-even
  return (ushort_t)(u >> 16);
}

__device__ __forceinline__ float gelu_tanh(float x) {
  float x3 = x * x * x;
  return 0.5f * x * (1.f + tanhf(0.7978845608028654f * (x + 0.044715f * x3)));
}

#if __has_builtin(__builtin_amdgcn_tensor_load_to_lds)
#define HAVE_TDM 1
#else
#define HAVE_TDM 0
#endif

#if HAVE_TDM
// 2-D TDM tile load -> LDS with dword padding (conflict-free layout).
// dsz_code: 0=1B,1=2B,2=4B ; pad_code: interval 2^(code+1) dwords, +1 dword pad.
__device__ __forceinline__ void tdm_load_2d(unsigned lds_off, const void* gptr,
                                            unsigned tile_d0, unsigned tile_d1,
                                            unsigned stride0, unsigned dsz_code,
                                            unsigned pad_code) {
  unsigned long long ga = (unsigned long long)gptr;
  v4u g0;
  g0[0] = 1u;                                   // count=1, user mode, no gather
  g0[1] = lds_off;                              // lds_addr (bytes)
  g0[2] = (unsigned)(ga & 0xFFFFFFFFu);         // global_addr lo
  g0[3] = (unsigned)((ga >> 32) & 0x01FFFFFFu)  // global_addr hi (57-bit)
          | (2u << 30);                         // type = 2
  v8i g1;
  g1[0] = (int)((dsz_code << 16) | (1u << 20) | (pad_code << 22));
  g1[1] = (int)((tile_d0 & 0xFFFFu) << 16);     // tensor_dim0 = tile_d0
  g1[2] = (int)(((tile_d0 >> 16) & 0xFFFFu) | ((tile_d1 & 0xFFFFu) << 16));
  g1[3] = (int)(((tile_d1 >> 16) & 0xFFFFu) | ((tile_d0 & 0xFFFFu) << 16)); // tile_dim0
  g1[4] = (int)(tile_d1 & 0xFFFFu);             // tile_dim1; tile_dim2=0
  g1[5] = (int)stride0;                         // tensor_dim0_stride lo32
  g1[6] = 0;
  g1[7] = 0;
  v4i z4 = {0, 0, 0, 0};
#if defined(__clang_major__) && (__clang_major__ >= 23)
  v8i z8 = {0, 0, 0, 0, 0, 0, 0, 0};
  __builtin_amdgcn_tensor_load_to_lds(g0, g1, z4, z4, z8, 0);
#else
  __builtin_amdgcn_tensor_load_to_lds(g0, g1, z4, z4, 0);
#endif
}
#endif

// fragment load: packed bf16 pairs as dwords, per the 16-bit A-matrix layout
__device__ __forceinline__ v16bf frag_ld(const ushort_t* rowp, int half) {
  const unsigned int* d = (const unsigned int*)rowp;
  v8u u;
#pragma unroll
  for (int i = 0; i < 8; ++i) {
    const int dk = ((i >> 2) * 8) + half * 4 + (i & 3);
    u[i] = d[dk];
  }
  return __builtin_bit_cast(v16bf, u);
}

// ---------------------------------------------------------------- WMMA GEMM (NT, bf16)
// C[M,N] = alpha * A[M,K] @ Bt[N,K]^T (+ bias[n]) (+ GELU)
// OUTK: 0 = fp32, 1 = bf16, 2 = bf16 transposed (C[n*ldc + m])
// M,N % 128 == 0, K % 32 == 0. Double-buffered TDM pipeline.
template <int OUTK, bool GELU>
__global__ __launch_bounds__(256) void gemm_bf16_nt(
    const ushort_t* __restrict__ A, const ushort_t* __restrict__ Bt,
    const float* __restrict__ bias, void* __restrict__ Cout,
    int M, int N, int K, int lda, int ldb, int ldc, float alpha) {
  __shared__ __align__(16) ushort_t As[2][TM * ALDS_E];
  __shared__ __align__(16) ushort_t Bs[2][TN * ALDS_E];

  const int tid  = threadIdx.x;
  const int lane = tid & 31;
  const int wave = tid >> 5;           // 8 waves
  const int wm   = wave & 3;           // 4 wave-rows (32 M each)
  const int wn   = wave >> 2;          // 2 wave-cols (64 N each)
  const int half = lane >> 4;
  const int l16  = lane & 15;
  const int row0 = blockIdx.y * TM;
  const int col0 = blockIdx.x * TN;

  v8f acc[2][4];
#pragma unroll
  for (int j = 0; j < 2; ++j)
#pragma unroll
    for (int t = 0; t < 4; ++t)
#pragma unroll
      for (int i = 0; i < 8; ++i) acc[j][t][i] = 0.f;

  const int nk = K / KSTEP;

  auto compute_from = [&](int b) {
    v16bf afr[2], bfr[4];
#pragma unroll
    for (int j = 0; j < 2; ++j)
      afr[j] = frag_ld(&As[b][(wm * 32 + j * 16 + l16) * ALDS_E], half);
#pragma unroll
    for (int t = 0; t < 4; ++t)
      bfr[t] = frag_ld(&Bs[b][(wn * 64 + t * 16 + l16) * ALDS_E], half);
#pragma unroll
    for (int j = 0; j < 2; ++j)
#pragma unroll
      for (int t = 0; t < 4; ++t)
        acc[j][t] = __builtin_amdgcn_wmma_f32_16x16x32_bf16(
            false, afr[j], false, bfr[t], (short)0, acc[j][t], false, false);
  };

#if HAVE_TDM
  unsigned as_off[2], bs_off[2];
#pragma unroll
  for (int b = 0; b < 2; ++b) {
    as_off[b] = (unsigned)(unsigned long long)(void*)&As[b][0];
    bs_off[b] = (unsigned)(unsigned long long)(void*)&Bs[b][0];
  }
  if (wave == 0) {   // prologue: k-step 0 into buffer 0
    tdm_load_2d(as_off[0], A + (long)row0 * lda, KSTEP, TM, (unsigned)lda, 1u, 3u);
    tdm_load_2d(bs_off[0], Bt + (long)col0 * ldb, KSTEP, TN, (unsigned)ldb, 1u, 3u);
  }
  for (int kt = 0; kt < nk; ++kt) {
    const int cur = kt & 1;
    if (wave == 0) {
      if (kt + 1 < nk) {   // issue next tile pair into the other buffer
        const int k1 = (kt + 1) * KSTEP;
        tdm_load_2d(as_off[cur ^ 1], A + (long)row0 * lda + k1, KSTEP, TM, (unsigned)lda, 1u, 3u);
        tdm_load_2d(bs_off[cur ^ 1], Bt + (long)col0 * ldb + k1, KSTEP, TN, (unsigned)ldb, 1u, 3u);
        __builtin_amdgcn_s_wait_tensorcnt(2);   // current pair complete
      } else {
        __builtin_amdgcn_s_wait_tensorcnt(0);
      }
    }
    __syncthreads();           // publish buf[cur]
    compute_from(cur);
    __syncthreads();           // buf[cur] free for k-step kt+2
  }
#else
  for (int kt = 0; kt < nk; ++kt) {
    const int k0 = kt * KSTEP;
    const int r = tid >> 1, c = (tid & 1) * 16;
    const ushort_t* ap = A + (long)(row0 + r) * lda + k0 + c;
    const ushort_t* bp = Bt + (long)(col0 + r) * ldb + k0 + c;
#pragma unroll
    for (int i = 0; i < 16; ++i) As[0][r * ALDS_E + c + i] = ap[i];
#pragma unroll
    for (int i = 0; i < 16; ++i) Bs[0][r * ALDS_E + c + i] = bp[i];
    __syncthreads();
    compute_from(0);
    __syncthreads();
  }
#endif

  // D layout: VGPR r -> m = r + 8*half, n = lane&15
  float* Cf = (float*)Cout;
  ushort_t* Cb = (ushort_t*)Cout;
#pragma unroll
  for (int j = 0; j < 2; ++j) {
    const int orow = row0 + wm * 32 + j * 16 + 8 * half;
#pragma unroll
    for (int t = 0; t < 4; ++t) {
      const int ocol = col0 + wn * 64 + t * 16 + l16;
      const float bv = bias ? bias[ocol] : 0.f;
#pragma unroll
      for (int r = 0; r < 8; ++r) {
        float v = alpha * acc[j][t][r] + bv;
        if (GELU) v = gelu_tanh(v);
        const long rr = orow + r;
        if (OUTK == 0)      Cf[rr * (long)ldc + ocol] = v;
        else if (OUTK == 1) Cb[rr * (long)ldc + ocol] = f2bf_us(v);
        else                Cb[(long)ocol * ldc + rr] = f2bf_us(v);
      }
    }
  }
}

// ---------------------------------------------------------------- elementwise
__global__ void addv(const float* __restrict__ a, const float* __restrict__ b,
                     float* __restrict__ o, int n) {
  int i = blockIdx.x * blockDim.x + threadIdx.x;
  if (i < n) o[i] = a[i] + b[i];
}

// fp32 -> bf16 flat convert
__global__ void conv_bf(const float* __restrict__ in, ushort_t* __restrict__ o, long n) {
  long i = blockIdx.x * (long)blockDim.x + threadIdx.x;
  if (i < n) o[i] = f2bf_us(in[i]);
}

// Wt[n*K + k] = bf16(W[k*N + n])  -- 32x32 LDS tile transpose
__global__ __launch_bounds__(256) void wtrans(const float* __restrict__ W,
                                              ushort_t* __restrict__ Wt,
                                              int K, int N) {
  __shared__ float t[32][33];
  const int kb = blockIdx.y * 32, nb = blockIdx.x * 32;
  const int tx = threadIdx.x & 31, ty = threadIdx.x >> 5;
  for (int i = ty; i < 32; i += 8) t[i][tx] = W[(long)(kb + i) * N + nb + tx];
  __syncthreads();
  for (int i = ty; i < 32; i += 8) Wt[(long)(nb + i) * K + kb + tx] = f2bf_us(t[tx][i]);
}

// out(bf16) = LN(x) * (add_one + w[c]) + b[c]
__global__ __launch_bounds__(256) void ln_affine(
    const float* __restrict__ x, ushort_t* __restrict__ out,
    const float* __restrict__ w, const float* __restrict__ b, float add_one,
    int C_) {
  __shared__ float r1[256], r2[256];
  const long row = blockIdx.x;
  const float* p = x + row * (long)C_;
  float s = 0.f, s2 = 0.f;
  for (int c = threadIdx.x; c < C_; c += 256) { float v = p[c]; s += v; s2 += v * v; }
  r1[threadIdx.x] = s; r2[threadIdx.x] = s2; __syncthreads();
  for (int o = 128; o > 0; o >>= 1) {
    if (threadIdx.x < o) { r1[threadIdx.x] += r1[threadIdx.x + o]; r2[threadIdx.x] += r2[threadIdx.x + o]; }
    __syncthreads();
  }
  const float mean = r1[0] / C_;
  const float var  = r2[0] / C_ - mean * mean;
  const float inv  = rsqrtf(var + 1e-6f);
  for (int c = threadIdx.x; c < C_; c += 256) {
    float nv = (p[c] - mean) * inv;
    out[row * (long)C_ + c] = f2bf_us(nv * (add_one + w[c]) + b[c]);
  }
}

// x = x * rsqrt(mean(x^2)+eps) * n[c]   (fp32 in place)
__global__ __launch_bounds__(256) void rms_rows(float* __restrict__ x,
                                                const float* __restrict__ n,
                                                int C_) {
  __shared__ float r[256];
  const long row = blockIdx.x;
  float* p = x + row * (long)C_;
  float s = 0.f;
  for (int c = threadIdx.x; c < C_; c += 256) { float v = p[c]; s += v * v; }
  r[threadIdx.x] = s; __syncthreads();
  for (int o = 128; o > 0; o >>= 1) {
    if (threadIdx.x < o) r[threadIdx.x] += r[threadIdx.x + o];
    __syncthreads();
  }
  const float inv = rsqrtf(r[0] / C_ + 1e-6f);
  for (int c = threadIdx.x; c < C_; c += 256) p[c] = p[c] * inv * n[c];
}

// cos/sin tables, 3-axis RoPE; F,H,W read from device scalars
__global__ void rope_tab(float* __restrict__ cosb, float* __restrict__ sinb,
                         const int* fp, const int* hp, const int* wp, int halfD) {
  const int H = *hp, W = *wp; (void)fp;
  const int l = blockIdx.x, j = threadIdx.x;      // blockDim.x == halfD
  const int c2 = halfD / 3;
  const int c1 = halfD - 2 * c2;
  const int fi = l / (H * W);
  const int rem = l % (H * W);
  const int hi = rem / W, wi = rem % W;
  float pos = (j < c1) ? (float)fi : ((j < c1 + c2) ? (float)hi : (float)wi);
  float inv = powf(10000.f, -(float)j / (float)halfD);
  float a = pos * inv;
  cosb[(long)l * halfD + j] = cosf(a);
  sinb[(long)l * halfD + j] = sinf(a);
}

__global__ void rope_apply(float* __restrict__ q, const float* __restrict__ cosb,
                           const float* __restrict__ sinb, int L_, int C_, int D_) {
  long i = blockIdx.x * (long)blockDim.x + threadIdx.x;
  const int halfD = D_ / 2, halfC = C_ / 2;
  const long total = (long)L_ * halfC;
  if (i >= total) return;
  const long l = i / halfC;
  const int r = (int)(i % halfC);
  const int h = r / halfD, j = r % halfD;
  const float co = cosb[l * halfD + j], si = sinb[l * halfD + j];
  const long base = l * (long)C_ + (long)h * D_ + 2 * j;
  const float x0 = q[base], x1 = q[base + 1];
  q[base]     = x0 * co - x1 * si;
  q[base + 1] = x0 * si + x1 * co;
}

// softmax: fp32 scores in (stride ld_in) -> bf16 probs out (stride ld_out)
__global__ __launch_bounds__(256) void softmax_bf(const float* __restrict__ s,
                                                  ushort_t* __restrict__ p,
                                                  int ncols, int ld_in, int ld_out) {
  __shared__ float red[256];
  const float* row = s + (long)blockIdx.x * ld_in;
  ushort_t* po = p + (long)blockIdx.x * ld_out;
  float m = -1e30f;
  for (int c = threadIdx.x; c < ncols; c += 256) m = fmaxf(m, row[c]);
  red[threadIdx.x] = m; __syncthreads();
  for (int o = 128; o > 0; o >>= 1) {
    if (threadIdx.x < o) red[threadIdx.x] = fmaxf(red[threadIdx.x], red[threadIdx.x + o]);
    __syncthreads();
  }
  m = red[0]; __syncthreads();
  float sum = 0.f;
  for (int c = threadIdx.x; c < ncols; c += 256) sum += expf(row[c] - m);
  red[threadIdx.x] = sum; __syncthreads();
  for (int o = 128; o > 0; o >>= 1) {
    if (threadIdx.x < o) red[threadIdx.x] += red[threadIdx.x + o];
    __syncthreads();
  }
  const float inv = 1.f / red[0];
  for (int c = threadIdx.x; c < ncols; c += 256)
    po[c] = f2bf_us(expf(row[c] - m) * inv);
}

// out[i] = base[i] + y[i] * (s ? s[i % C] : 1)
__global__ void resid(float* __restrict__ out, const float* __restrict__ base,
                      const float* __restrict__ y, const float* __restrict__ s,
                      long n, int C_) {
  long i = blockIdx.x * (long)blockDim.x + threadIdx.x;
  if (i >= n) return;
  const float sc = s ? s[i % C_] : 1.f;
  out[i] = base[i] + y[i] * sc;
}

// ---------------------------------------------------------------- launcher
extern "C" void kernel_launch(void* const* d_in, const int* in_sizes, int n_in,
                              void* d_out, int out_size, void* d_ws, size_t ws_size,
                              hipStream_t stream) {
  (void)n_in; (void)out_size; (void)ws_size;
  const float* x    = (const float*)d_in[0];
  const float* e    = (const float*)d_in[1];
  const float* ctx  = (const float*)d_in[2];
  const float* modv = (const float*)d_in[3];
  const float* sa_wq = (const float*)d_in[4];  const float* sa_bq = (const float*)d_in[5];
  const float* sa_wk = (const float*)d_in[6];  const float* sa_bk = (const float*)d_in[7];
  const float* sa_wv = (const float*)d_in[8];  const float* sa_bv = (const float*)d_in[9];
  const float* sa_wo = (const float*)d_in[10]; const float* sa_bo = (const float*)d_in[11];
  const float* sa_nq = (const float*)d_in[12]; const float* sa_nk = (const float*)d_in[13];
  const float* n3_w  = (const float*)d_in[14]; const float* n3_b  = (const float*)d_in[15];
  const float* ca_wq = (const float*)d_in[16]; const float* ca_bq = (const float*)d_in[17];
  const float* ca_wk = (const float*)d_in[18]; const float* ca_bk = (const float*)d_in[19];
  const float* ca_wv = (const float*)d_in[20]; const float* ca_bv = (const float*)d_in[21];
  const float* ca_wo = (const float*)d_in[22]; const float* ca_bo = (const float*)d_in[23];
  const float* ca_nq = (const float*)d_in[24]; const float* ca_nk = (const float*)d_in[25];
  const float* ffn_w1 = (const float*)d_in[26]; const float* ffn_b1 = (const float*)d_in[27];
  const float* ffn_w2 = (const float*)d_in[28]; const float* ffn_b2 = (const float*)d_in[29];
  const int* fp = (const int*)d_in[30];
  const int* hp = (const int*)d_in[31];
  const int* wp = (const int*)d_in[32];
  float* out = (float*)d_out;

  const int C    = in_sizes[1] / 6;           // 1536
  const int L    = in_sizes[0] / C;           // 4032
  const int L2   = in_sizes[2] / C;           // 512
  const int FFN  = in_sizes[27];              // 8960
  const int NHh  = 12, D = C / NHh, halfD = D / 2;
  const int Lp   = ((L + TM - 1) / TM) * TM;  // 4096 (padded rows/cols)
  const float scale = 1.f / sqrtf((float)D);

  // ---- workspace carve (bytes, 256B aligned)
  char* wsb = (char*)d_ws;
  size_t off = 0;
  auto carveB = [&](size_t bytes) { void* p = wsb + off; off += (bytes + 255) & ~(size_t)255; return p; };
  // fp32 buffers
  float* em     = (float*)carveB((size_t)6 * C * 4);
  float* cosb   = (float*)carveB((size_t)L * halfD * 4);
  float* sinb   = (float*)carveB((size_t)L * halfD * 4);
  float* qf     = (float*)carveB((size_t)Lp * C * 4);
  float* kf     = (float*)carveB((size_t)Lp * C * 4);
  float* tmpf   = (float*)carveB((size_t)Lp * C * 4);
  float* scores = (float*)carveB((size_t)Lp * Lp * 4);
  // bf16 buffers
  ushort_t* xab  = (ushort_t*)carveB((size_t)Lp * C * 2);
  ushort_t* qbh  = (ushort_t*)carveB((size_t)Lp * C * 2);
  ushort_t* kbh  = (ushort_t*)carveB((size_t)Lp * C * 2);
  ushort_t* vt   = (ushort_t*)carveB((size_t)C * Lp * 2);   // [C][Lp] transposed V
  ushort_t* yb   = (ushort_t*)carveB((size_t)Lp * C * 2);
  ushort_t* prob = (ushort_t*)carveB((size_t)Lp * Lp * 2);
  ushort_t* hid  = (ushort_t*)carveB((size_t)Lp * FFN * 2);
  ushort_t* ctxb = (ushort_t*)carveB((size_t)L2 * C * 2);
  // bf16 transposed weights [N][K]
  ushort_t* wT[10];
  const float* wsrc[10] = {sa_wq, sa_wk, sa_wv, sa_wo, ca_wq, ca_wk, ca_wv, ca_wo, ffn_w1, ffn_w2};
  const int    wK[10]   = {C, C, C, C, C, C, C, C, C, FFN};
  const int    wN[10]   = {C, C, C, C, C, C, C, C, FFN, C};
  for (int i = 0; i < 10; ++i) wT[i] = (ushort_t*)carveB((size_t)wK[i] * wN[i] * 2);

  auto G = [&](int outk, bool gel, const ushort_t* A, int lda, const ushort_t* Bt,
               int ldb, const float* bias, void* Cm, int ldc, int M, int N, int K,
               float alpha) {
    dim3 g(N / TN, M / TM);
    if (outk == 0)
      gemm_bf16_nt<0, false><<<g, 256, 0, stream>>>(A, Bt, bias, Cm, M, N, K, lda, ldb, ldc, alpha);
    else if (outk == 1 && gel)
      gemm_bf16_nt<1, true><<<g, 256, 0, stream>>>(A, Bt, bias, Cm, M, N, K, lda, ldb, ldc, alpha);
    else if (outk == 1)
      gemm_bf16_nt<1, false><<<g, 256, 0, stream>>>(A, Bt, bias, Cm, M, N, K, lda, ldb, ldc, alpha);
    else
      gemm_bf16_nt<2, false><<<g, 256, 0, stream>>>(A, Bt, bias, Cm, M, N, K, lda, ldb, ldc, alpha);
  };
  auto CONV = [&](const float* in, ushort_t* o, long n) {
    conv_bf<<<(unsigned)((n + 255) / 256), 256, 0, stream>>>(in, o, n);
  };

  const long LC = (long)L * C;

  // ---- weight transpose+convert (fp32 [K,N] -> bf16 [N,K]), once per call
  for (int i = 0; i < 10; ++i)
    wtrans<<<dim3(wN[i] / 32, wK[i] / 32), 256, 0, stream>>>(wsrc[i], wT[i], wK[i], wN[i]);
  CONV(ctx, ctxb, (long)L2 * C);
  addv<<<(6 * C + 255) / 256, 256, 0, stream>>>(modv, e, em, 6 * C);

  // ================= self-attention =================
  ln_affine<<<L, 256, 0, stream>>>(x, xab, em + C, em, 1.f, C);
  G(0, false, xab, C, wT[0], C, sa_bq, qf, C, Lp, C, C, 1.f);
  G(0, false, xab, C, wT[1], C, sa_bk, kf, C, Lp, C, C, 1.f);
  G(2, false, xab, C, wT[2], C, sa_bv, vt, Lp, Lp, C, C, 1.f);   // V, bf16-transposed
  rms_rows<<<L, 256, 0, stream>>>(qf, sa_nq, C);
  rms_rows<<<L, 256, 0, stream>>>(kf, sa_nk, C);
  rope_tab<<<L, halfD, 0, stream>>>(cosb, sinb, fp, hp, wp, halfD);
  {
    long tot = (long)L * (C / 2);
    rope_apply<<<(unsigned)((tot + 255) / 256), 256, 0, stream>>>(qf, cosb, sinb, L, C, D);
    rope_apply<<<(unsigned)((tot + 255) / 256), 256, 0, stream>>>(kf, cosb, sinb, L, C, D);
  }
  CONV(qf, qbh, (long)Lp * C);
  CONV(kf, kbh, (long)Lp * C);
  for (int h = 0; h < NHh; ++h) {
    G(0, false, qbh + h * D, C, kbh + h * D, C, nullptr, scores, Lp, Lp, Lp, D, scale);
    softmax_bf<<<L, 256, 0, stream>>>(scores, prob, L, Lp, Lp);
    G(1, false, prob, Lp, vt + (long)h * D * Lp, Lp, nullptr, yb + h * D, C, Lp, D, L, 1.f);
  }
  G(0, false, yb, C, wT[3], C, sa_bo, tmpf, C, Lp, C, C, 1.f);
  resid<<<(unsigned)((LC + 255) / 256), 256, 0, stream>>>(out, x, tmpf, em + 2 * C, LC, C);

  // ================= cross-attention =================
  ln_affine<<<L, 256, 0, stream>>>(out, xab, n3_w, n3_b, 0.f, C);
  G(0, false, xab, C, wT[4], C, ca_bq, qf, C, Lp, C, C, 1.f);
  rms_rows<<<L, 256, 0, stream>>>(qf, ca_nq, C);
  CONV(qf, qbh, (long)Lp * C);
  G(0, false, ctxb, C, wT[5], C, ca_bk, kf, C, L2, C, C, 1.f);
  rms_rows<<<L2, 256, 0, stream>>>(kf, ca_nk, C);
  CONV(kf, kbh, (long)L2 * C);
  G(2, false, ctxb, C, wT[6], C, ca_bv, vt, Lp, L2, C, C, 1.f);
  for (int h = 0; h < NHh; ++h) {
    G(0, false, qbh + h * D, C, kbh + h * D, C, nullptr, scores, L2, Lp, L2, D, scale);
    softmax_bf<<<L, 256, 0, stream>>>(scores, prob, L2, L2, L2);
    G(1, false, prob, L2, vt + (long)h * D * Lp, Lp, nullptr, yb + h * D, C, Lp, D, L2, 1.f);
  }
  G(0, false, yb, C, wT[7], C, ca_bo, tmpf, C, Lp, C, C, 1.f);
  resid<<<(unsigned)((LC + 255) / 256), 256, 0, stream>>>(out, out, tmpf, nullptr, LC, C);

  // ================= FFN =================
  ln_affine<<<L, 256, 0, stream>>>(out, xab, em + 4 * C, em + 3 * C, 1.f, C);
  G(1, true, xab, C, wT[8], C, ffn_b1, hid, FFN, Lp, FFN, C, 1.f);   // fused GELU, bf16 out
  G(0, false, hid, FFN, wT[9], FFN, ffn_b2, tmpf, C, Lp, C, FFN, 1.f);
  resid<<<(unsigned)((LC + 255) / 256), 256, 0, stream>>>(out, out, tmpf, em + 5 * C, LC, C);
}
